// ForcedAligner_35673998361283
// MI455X (gfx1250) — matrix operand order repaired and checked by reading, not compiled, not run on hardware
//
#include <hip/hip_runtime.h>
#include <hip/hip_bf16.h>

// MI455X / gfx1250, wave32. All GEMM-shaped work via v_wmma_f32_16x16x32_bf16
// (f32 accumulate). Intermediate activations and pre-transposed weights are
// stored bf16 to halve HBM traffic; the 134MB output is written exactly once.

typedef __attribute__((ext_vector_type(16))) __bf16 v16bf;
typedef __attribute__((ext_vector_type(8)))  float  v8f;

#define ADIM 128
#define ODIM 80
#define BATCH 8
#define T_TEXT 1024
#define T_FEATS 4096

// ---------------------------------------------------------------------------
// Weight prep: (Cout=128, Cin, K) f32 -> wT[cout][k] bf16, k = tap*Cin + ci,
// zero-padded to Kpad (multiple of 32). Makes conv B-fragments 16 contiguous
// bf16 per lane.
// ---------------------------------------------------------------------------
__global__ void prep_weights_kernel(const float* __restrict__ w,
                                    __bf16* __restrict__ wT,
                                    int Cin, int ktaps, int Kpad) {
  int idx = blockIdx.x * 256 + threadIdx.x;
  if (idx >= 128 * Kpad) return;
  int n = idx / Kpad;
  int k = idx - n * Kpad;
  float v = 0.0f;
  if (k < Cin * ktaps) {
    int tap = k / Cin;
    int ci  = k - tap * Cin;
    v = w[((size_t)n * Cin + ci) * ktaps + tap];
  }
  wT[idx] = (__bf16)v;
}

// ---------------------------------------------------------------------------
// Conv1d as WMMA GEMM. WG: 16 positions x 128 out-channels (8 waves x 16).
// K = KTAPS*CIN (tap-major), compile-time -> fully unrolled, no div/mod.
// IN_LAYOUT: 0 -> in[b][ci][t] f32 (raw inputs), 1 -> in[b][t][ci] bf16.
// Output: out[b][t][cout] bf16.
// ---------------------------------------------------------------------------
template <int RELU, int IN_LAYOUT, int CIN, int KTAPS, typename TIN>
__global__ void conv1d_wmma_kernel(const TIN* __restrict__ in,
                                   const __bf16* __restrict__ wT,  // [128][Kpad]
                                   const float* __restrict__ bias,
                                   __bf16* __restrict__ out,
                                   int T) {
  constexpr int PAD    = (KTAPS == 3) ? 1 : 0;
  constexpr int POS    = 16 + 2 * PAD;
  constexpr int KTOT   = CIN * KTAPS;
  constexpr int NSTEPS = (KTOT + 31) >> 5;
  constexpr int KPAD   = NSTEPS * 32;

  const int t0   = blockIdx.x * 16;
  const int bb   = blockIdx.y;
  const int tid  = threadIdx.x;
  const int wave = tid >> 5;
  const int lane = tid & 31;
  const int hf   = lane >> 4;
  const int nl   = lane & 15;

  __shared__ __bf16 s_in[POS * CIN];   // [pos][ci]

  for (int idx = tid; idx < POS * CIN; idx += 256) {
    int p  = idx / CIN;
    int ci = idx - p * CIN;
    int tp = t0 + p - PAD;
    float v = 0.0f;
    if (tp >= 0 && tp < T) {
      v = (IN_LAYOUT == 0) ? (float)in[((size_t)bb * CIN + ci) * T + tp]
                           : (float)in[((size_t)bb * T + tp) * CIN + ci];
    }
    s_in[idx] = (__bf16)v;
  }
  __syncthreads();

  v8f acc;
#pragma unroll
  for (int j = 0; j < 8; ++j) acc[j] = 0.0f;

  const int n = wave * 16 + nl;                       // out channel
  const __bf16* wrow = wT + (size_t)n * KPAD;

#pragma unroll
  for (int s = 0; s < NSTEPS; ++s) {
    const int kb = s << 5;

    // A fragment: A[m][k] = s_in[(m+tap)*CIN + ci], m = nl
    v16bf afr;
#pragma unroll
    for (int v = 0; v < 8; ++v) {
#pragma unroll
      for (int j = 0; j < 2; ++j) {
        int kk = hf * 16 + v * 2 + j;
        int ki = kb + kk;
        __bf16 bv = (__bf16)0.0f;
        if (ki < KTOT) {
          int tap = ki / CIN;            // compile-time divisor
          int ci  = ki - tap * CIN;
          bv = s_in[(nl + tap) * CIN + ci];
        }
        afr[2 * v + j] = bv;
      }
    }

    // B fragment: per lane = column n, 16 contiguous bf16 from wT
    const __bf16* wk = wrow + kb + hf * 16;
    v16bf bfr;
#pragma unroll
    for (int e = 0; e < 16; ++e) bfr[e] = wk[e];

    acc = __builtin_amdgcn_wmma_f32_16x16x32_bf16(
        false, afr, false, bfr, (short)0, acc, false, false);
  }

  const float bv = bias[n];
#pragma unroll
  for (int j = 0; j < 8; ++j) {
    int mr = j + 8 * hf;                 // C layout: VGPR j -> M=j / j+8
    float v = acc[j] + bv;
    if (RELU) v = fmaxf(v, 0.0f);
    out[((size_t)bb * T + (t0 + mr)) * ADIM + n] = (__bf16)v;
  }
}

// ---------------------------------------------------------------------------
// Row sum-of-squares over 128 bf16 channels: one wave per row.
// ---------------------------------------------------------------------------
__global__ void rowsq_kernel(const __bf16* __restrict__ x,
                             float* __restrict__ sq, int rows) {
  int row  = blockIdx.x * 8 + (threadIdx.x >> 5);
  int lane = threadIdx.x & 31;
  if (row >= rows) return;
  const __bf16* r = x + (size_t)row * ADIM;
  float s = 0.0f;
#pragma unroll
  for (int e = 0; e < 4; ++e) {
    float v = (float)r[lane * 4 + e];
    s += v * v;
  }
#pragma unroll
  for (int off = 16; off; off >>= 1) s += __shfl_xor(s, off, 32);
  if (lane == 0) sq[row] = s;
}

// ---------------------------------------------------------------------------
// Fused: cross = f @ t^T (M=16 feat rows, N=1024 text cols, K=128)
//        -> sq-dist -> -sqrt -> mask -> row log-softmax -> single store.
// Wave w owns cols [w*128, w*128+128) as 8 WMMA tiles (64 VGPRs of C).
// t_act (256KB/batch, bf16) stays L2-resident across 256 WGs per batch.
// ---------------------------------------------------------------------------
__global__ void cross_softmax_kernel(const __bf16* __restrict__ f_act,
                                     const __bf16* __restrict__ t_act,
                                     const float* __restrict__ f_sq,
                                     const float* __restrict__ t_sq,
                                     const unsigned char* __restrict__ masks,
                                     float* __restrict__ out) {
  const int f0   = blockIdx.x * 16;
  const int tb   = blockIdx.y;
  const int tid  = threadIdx.x;
  const int wave = tid >> 5;
  const int lane = tid & 31;
  const int hf   = lane >> 4;
  const int nl   = lane & 15;

  __shared__ __bf16 s_f[16 * ADIM];
  __shared__ float  s_fsq[16];
  __shared__ float  s_redmax[16][8];
  __shared__ float  s_redsum[16][8];

  for (int idx = tid; idx < 16 * ADIM; idx += 256) {
    int r = idx >> 7, c = idx & 127;
    s_f[idx] = f_act[((size_t)tb * T_FEATS + f0 + r) * ADIM + c];
  }
  if (tid < 16) s_fsq[tid] = f_sq[(size_t)tb * T_FEATS + f0 + tid];
  __syncthreads();

  v8f acc[8];
#pragma unroll
  for (int i = 0; i < 8; ++i)
#pragma unroll
    for (int j = 0; j < 8; ++j) acc[i][j] = 0.0f;

#pragma unroll
  for (int s = 0; s < 4; ++s) {
    const int kb = s << 5;
    v16bf afr;
#pragma unroll
    for (int v = 0; v < 8; ++v) {
#pragma unroll
      for (int j = 0; j < 2; ++j) {
        int kk = hf * 16 + v * 2 + j;
        afr[2 * v + j] = s_f[nl * ADIM + kb + kk];
      }
    }
#pragma unroll
    for (int i = 0; i < 8; ++i) {
      const int cb = wave * 128 + i * 16;
      const __bf16* trow =
          t_act + ((size_t)tb * T_TEXT + cb + nl) * ADIM + kb + hf * 16;
      v16bf bfr;
#pragma unroll
      for (int e = 0; e < 16; ++e) bfr[e] = trow[e];
      acc[i] = __builtin_amdgcn_wmma_f32_16x16x32_bf16(
          false, afr, false, bfr, (short)0, acc[i], false, false);
    }
  }

  // Epilogue: score = -sqrt(max(fsq + tsq - 2*cross, 0)); mask -> -inf.
#pragma unroll
  for (int i = 0; i < 8; ++i) {
    const int ncol = wave * 128 + i * 16 + nl;
    const float ts = t_sq[(size_t)tb * T_TEXT + ncol];
    const bool mk  = masks[(size_t)tb * T_TEXT + ncol] != 0;
#pragma unroll
    for (int j = 0; j < 8; ++j) {
      int mr = j + 8 * hf;
      float q = s_fsq[mr] + ts - 2.0f * acc[i][j];
      float v = -sqrtf(fmaxf(q, 0.0f));
      acc[i][j] = mk ? -__builtin_inff() : v;
    }
  }

  // Row max: 8 tiles locally, shuffle across the 16-lane half, LDS over waves.
  float rmax[8];
#pragma unroll
  for (int j = 0; j < 8; ++j) {
    float m = acc[0][j];
#pragma unroll
    for (int i = 1; i < 8; ++i) m = fmaxf(m, acc[i][j]);
#pragma unroll
    for (int off = 1; off < 16; off <<= 1) m = fmaxf(m, __shfl_xor(m, off, 32));
    if (nl == 0) s_redmax[j + 8 * hf][wave] = m;
  }
  __syncthreads();
#pragma unroll
  for (int j = 0; j < 8; ++j) {
    int mr = j + 8 * hf;
    float m = s_redmax[mr][0];
#pragma unroll
    for (int w2 = 1; w2 < 8; ++w2) m = fmaxf(m, s_redmax[mr][w2]);
    rmax[j] = m;
  }

  // Row sum(exp(score - max)).
#pragma unroll
  for (int j = 0; j < 8; ++j) {
    float s = 0.0f;
#pragma unroll
    for (int i = 0; i < 8; ++i) s += __expf(acc[i][j] - rmax[j]);
#pragma unroll
    for (int off = 1; off < 16; off <<= 1) s += __shfl_xor(s, off, 32);
    if (nl == 0) s_redsum[j + 8 * hf][wave] = s;
  }
  __syncthreads();

#pragma unroll
  for (int j = 0; j < 8; ++j) {
    int mr = j + 8 * hf;
    float s = 0.0f;
#pragma unroll
    for (int w2 = 0; w2 < 8; ++w2) s += s_redsum[mr][w2];
    float lz = rmax[j] + __logf(s);
    float* orow = out + ((size_t)tb * T_FEATS + f0 + mr) * T_TEXT;
#pragma unroll
    for (int i = 0; i < 8; ++i)
      orow[wave * 128 + i * 16 + nl] = acc[i][j] - lz;   // -inf stays -inf
  }
}

// ---------------------------------------------------------------------------
extern "C" void kernel_launch(void* const* d_in, const int* in_sizes, int n_in,
                              void* d_out, int out_size, void* d_ws, size_t ws_size,
                              hipStream_t stream) {
  (void)in_sizes; (void)n_in; (void)out_size; (void)ws_size;
  const float* text  = (const float*)d_in[0];
  const float* feats = (const float*)d_in[1];
  const unsigned char* x_masks = (const unsigned char*)d_in[2];  // numpy bool
  const float* t_w1 = (const float*)d_in[3];
  const float* t_b1 = (const float*)d_in[4];
  const float* t_w2 = (const float*)d_in[5];
  const float* t_b2 = (const float*)d_in[6];
  const float* f_w1 = (const float*)d_in[7];
  const float* f_b1 = (const float*)d_in[8];
  const float* f_w2 = (const float*)d_in[9];
  const float* f_b2 = (const float*)d_in[10];
  const float* f_w3 = (const float*)d_in[11];
  const float* f_b3 = (const float*)d_in[12];

  // Workspace carve-up (~29 MB), 256B-aligned chunks.
  char* p = (char*)d_ws;
  auto carve = [&p](size_t bytes) -> void* {
    void* r = (void*)p;
    p += (bytes + 255) & ~(size_t)255;
    return r;
  };
  __bf16* t1    = (__bf16*)carve((size_t)BATCH * T_TEXT  * ADIM * 2);
  __bf16* t_act = (__bf16*)carve((size_t)BATCH * T_TEXT  * ADIM * 2);
  __bf16* f1    = (__bf16*)carve((size_t)BATCH * T_FEATS * ADIM * 2);
  __bf16* f2    = (__bf16*)carve((size_t)BATCH * T_FEATS * ADIM * 2);
  __bf16* f_act = (__bf16*)carve((size_t)BATCH * T_FEATS * ADIM * 2);
  float*  t_sq  = (float*)carve((size_t)BATCH * T_TEXT  * 4);
  float*  f_sq  = (float*)carve((size_t)BATCH * T_FEATS * 4);
  __bf16* wt1   = (__bf16*)carve((size_t)128 * 384 * 2);   // Kpad 384
  __bf16* wt2   = (__bf16*)carve((size_t)128 * 128 * 2);   // Kpad 128
  __bf16* wf1   = (__bf16*)carve((size_t)128 * 256 * 2);   // Kpad 256 (240->256)
  __bf16* wf2   = (__bf16*)carve((size_t)128 * 384 * 2);   // Kpad 384
  __bf16* wf3   = (__bf16*)carve((size_t)128 * 128 * 2);   // Kpad 128

  dim3 blk(256);
  // Weight pre-transpose (tiny).
  prep_weights_kernel<<<(128 * 384 + 255) / 256, blk, 0, stream>>>(t_w1, wt1, ADIM, 3, 384);
  prep_weights_kernel<<<(128 * 128 + 255) / 256, blk, 0, stream>>>(t_w2, wt2, ADIM, 1, 128);
  prep_weights_kernel<<<(128 * 256 + 255) / 256, blk, 0, stream>>>(f_w1, wf1, ODIM, 3, 256);
  prep_weights_kernel<<<(128 * 384 + 255) / 256, blk, 0, stream>>>(f_w2, wf2, ADIM, 3, 384);
  prep_weights_kernel<<<(128 * 128 + 255) / 256, blk, 0, stream>>>(f_w3, wf3, ADIM, 1, 128);

  // Text branch: conv3+relu, conv1.
  conv1d_wmma_kernel<1, 0, ADIM, 3, float>
      <<<dim3(T_TEXT / 16, BATCH), blk, 0, stream>>>(text, wt1, t_b1, t1, T_TEXT);
  conv1d_wmma_kernel<0, 1, ADIM, 1, __bf16>
      <<<dim3(T_TEXT / 16, BATCH), blk, 0, stream>>>(t1, wt2, t_b2, t_act, T_TEXT);
  // Feats branch: conv3+relu, conv3+relu, conv1.
  conv1d_wmma_kernel<1, 0, ODIM, 3, float>
      <<<dim3(T_FEATS / 16, BATCH), blk, 0, stream>>>(feats, wf1, f_b1, f1, T_FEATS);
  conv1d_wmma_kernel<1, 1, ADIM, 3, __bf16>
      <<<dim3(T_FEATS / 16, BATCH), blk, 0, stream>>>(f1, wf2, f_b2, f2, T_FEATS);
  conv1d_wmma_kernel<0, 1, ADIM, 1, __bf16>
      <<<dim3(T_FEATS / 16, BATCH), blk, 0, stream>>>(f2, wf3, f_b3, f_act, T_FEATS);
  // Squared norms from the same bf16 vectors used in the Gram matmul.
  rowsq_kernel<<<(BATCH * T_TEXT) / 8, blk, 0, stream>>>(t_act, t_sq, BATCH * T_TEXT);
  rowsq_kernel<<<(BATCH * T_FEATS) / 8, blk, 0, stream>>>(f_act, f_sq, BATCH * T_FEATS);
  // Fused Gram + distance + mask + log-softmax, single pass over the output.
  cross_softmax_kernel<<<dim3(T_FEATS / 16, BATCH), blk, 0, stream>>>(
      f_act, t_act, f_sq, t_sq, x_masks, (float*)d_out);
}
